// SyntheticTreePropagationNetwork_89876485636312
// MI455X (gfx1250) — matrix operand order repaired and checked by reading, not compiled, not run on hardware
//
#include <hip/hip_runtime.h>
#include <hip/hip_bf16.h>

// Problem constants (from reference): B=512, L=64, H=256, OUT=2046+2=2048
#define BB  512
#define LL  64
#define HH  256
#define OO  2048

typedef __attribute__((ext_vector_type(2)))  float    v2f;
typedef __attribute__((ext_vector_type(8)))  float    v8f;
typedef __attribute__((ext_vector_type(8)))  _Float16 v8h;
typedef __attribute__((ext_vector_type(16))) _Float16 v16h;
typedef unsigned int uint32;
typedef __attribute__((ext_vector_type(4))) uint32 u32x4;
typedef __attribute__((ext_vector_type(8))) uint32 u32x8;

// ---------------------------------------------------------------------------
// Tensor Data Mover helpers (CDNA5). Inline asm is used instead of
// __builtin_amdgcn_tensor_load_to_lds because the two toolchains disagree on
// the builtin's arity; the ISA mnemonic takes D# group0 (4 SGPRs) and
// group1 (8 SGPRs) for tensors up to 2D. Completion is tracked by TENSORcnt.
// ---------------------------------------------------------------------------
__device__ __forceinline__ void tdm_wait0() {
#if __has_builtin(__builtin_amdgcn_s_wait_tensorcnt)
  __builtin_amdgcn_s_wait_tensorcnt(0);
#else
  asm volatile("s_wait_tensorcnt 0x0" ::: "memory");
#endif
}

// 2D tile DMA: global (row-major, row_stride elements between dim1 lines)
// -> LDS at lds_off, with optional per-row LDS padding (pad codes per D#:
// interval code v pads after 2^(v+1) dwords; amount code a pads a+1 dwords).
__device__ __forceinline__ void tdm_load_2d(
    uint32 lds_off, const void* gptr,
    uint32 tensor_d0, uint32 tensor_d1, uint32 tile_d0, uint32 tile_d1,
    uint32 row_stride, uint32 data_size_code,
    uint32 pad_en, uint32 pad_interval, uint32 pad_amount) {
  unsigned long long ga = (unsigned long long)gptr;
  u32x4 g0;
  g0.x = 1u;                                            // count=1, user mode
  g0.y = lds_off;                                       // lds_addr[31:0]
  g0.z = (uint32)ga;                                    // global_addr[31:0]
  g0.w = ((uint32)(ga >> 32) & 0x01FFFFFFu) | (2u << 30); // addr[56:32]|type=2
  u32x8 g1;
  g1[0] = (data_size_code << 16) | (pad_en << 20) |
          (pad_interval << 22) | (pad_amount << 25);    // wg_mask=0 (no cluster)
  g1[1] = (tensor_d0 & 0xFFFFu) << 16;                  // tensor_dim0[15:0]
  g1[2] = (tensor_d0 >> 16) | ((tensor_d1 & 0xFFFFu) << 16);
  g1[3] = (tensor_d1 >> 16) | (tile_d0 << 16);
  g1[4] = tile_d1 & 0xFFFFu;                            // tile_dim2 = 0
  g1[5] = row_stride;                                   // dim0_stride[31:0]
  g1[6] = 0;                                            // stride hi / dim1_stride
  g1[7] = 0;
  asm volatile("tensor_load_to_lds %0, %1" :: "s"(g0), "s"(g1) : "memory");
}

// ---------------------------------------------------------------------------
// Kernel 1: sibling-mean written directly into z (z[:,t] <- sib_mean[:,t],
// z[:,0] <- 0). One block per batch; the 64KB (64x256 f32) mol tile is DMA'd
// into LDS once by the TDM, so mol_vec is read exactly once from HBM.
// ---------------------------------------------------------------------------
__global__ __launch_bounds__(256) void sib_prep_kernel(
    const float* __restrict__ mol, const int* __restrict__ par,
    float* __restrict__ z) {
  __shared__ float smol[LL * HH];   // 64 KB
  __shared__ float ssum[LL * HH];   // 64 KB
  __shared__ int   spar[LL];
  __shared__ int   scnt[LL];
  const int b = blockIdx.x, tid = threadIdx.x;

  if ((tid >> 5) == 0) {            // wave 0 drives the DMA
    tdm_load_2d((uint32)(uintptr_t)smol, mol + (size_t)b * LL * HH,
                LL * HH, 1, LL * HH, 1, LL * HH, 2u /*4B*/, 0u, 0u, 0u);
    tdm_wait0();
  }
  if (tid < LL) { spar[tid] = par[b * LL + tid]; scnt[tid] = 0; }
  __syncthreads();
  if (tid >= 1 && tid < LL) atomicAdd(&scnt[spar[tid]], 1);
  for (int p = 0; p < LL; ++p) ssum[p * HH + tid] = 0.0f;
  __syncthreads();

  for (int t = 1; t < LL; ++t)      // per-thread channel: race-free LDS RMW
    ssum[spar[t] * HH + tid] += smol[t * HH + tid];
  __syncthreads();

  float* zb = z + (size_t)b * LL * HH;
  zb[tid] = 0.0f;                   // t = 0 row
  for (int t = 1; t < LL; ++t) {
    int   p = spar[t];
    int   c = scnt[p] - 1;          // siblings = children - self
    float v = 0.0f;
    if (c > 0) v = (ssum[p * HH + tid] - smol[t * HH + tid]) / (float)c;
    zb[t * HH + tid] = v;
  }
}

// ---------------------------------------------------------------------------
// Kernel 2: persistent tree recurrence. 32 blocks x 256 threads; each block
// owns 16 batches (deps are intra-batch, parent<t) and iterates t=1..63 with
// workgroup barriers only. Wp (f32) is TDM-DMA'd once into LDS with padded
// rows (stride 272 dwords via pad codes 7/15) and stays resident all steps.
// Per step: TDM streams the 16 sibling rows into LDS (overlapped with the
// waves' vector gather of parent rows), then V_WMMA_F32_16X16X4_F32.
// ---------------------------------------------------------------------------
#define WS 272   // Wp LDS row stride (dwords): pad 256+16
#define AS 260   // A/sib tile LDS row stride (dwords): pad 256+4

__global__ __launch_bounds__(256) void tree_recurrence_kernel(
    const float* __restrict__ Wp, const float* __restrict__ bp,
    const int* __restrict__ par, float* __restrict__ z) {
  __shared__ float wlds[HH * WS];   // 278,528 B
  __shared__ float sA[16 * AS];     //  16,640 B (gathered parent rows)
  __shared__ float sS[16 * AS];     //  16,640 B (sibling-mean rows)
                                    //  total 311,808 B < 320 KB WGP LDS
  const int tid  = threadIdx.x;
  const int m0   = blockIdx.x * 16;
  const int wave = tid >> 5;        // wave32
  const int lane = tid & 31;
  const int hi   = lane >> 4;
  const int lm   = lane & 15;
  const int n0a  = wave * 32;       // 2 N-tiles per wave -> N=256

  if (wave == 0) {                  // one-shot weight residency DMA
    // 256x256 f32 tile, LDS rows padded to 272 dwords (interval=7 -> every
    // 256 dwords, amount=15 -> 16 dwords).
    tdm_load_2d((uint32)(uintptr_t)wlds, Wp, HH, HH, HH, HH, HH,
                2u /*4B*/, 1u, 7u, 15u);
  }
  const float bias0 = bp[n0a + lm];
  const float bias1 = bp[n0a + 16 + lm];

  for (int t = 1; t < LL; ++t) {
    if (wave == 0) {
      // Sibling rows z[(m0+i)*64 + t], i=0..15: strided 2D tile (row stride
      // 64*256 elems), LDS rows padded to 260 dwords (interval=7, amount=3).
      tdm_load_2d((uint32)(uintptr_t)sS,
                  z + ((size_t)m0 * LL + t) * HH,
                  HH, 16, HH, 16, LL * HH, 2u /*4B*/, 1u, 7u, 3u);
    }
    // Meanwhile all waves vector-gather the 16 parent rows (p<t, final).
    for (int i = 0; i < 16; ++i) {
      int bb = m0 + i;
      int p  = par[bb * LL + t];
      sA[i * AS + tid] = z[((size_t)bb * LL + p) * HH + tid];
    }
    if (wave == 0) tdm_wait0();     // TENSORcnt==0 before signaling barrier
    __syncthreads();

    v8f acc0 = {}; v8f acc1 = {};
#pragma unroll 4
    for (int k = 0; k < HH; k += 4) {
      const int kk = k + 2 * hi;    // f32 A: lanes16-31 hold K=2,3
      v2f a;  a.x = sA[lm * AS + kk];  a.y = sA[lm * AS + kk + 1];
      v2f b0; b0.x = wlds[kk * WS + n0a + lm];
              b0.y = wlds[(kk + 1) * WS + n0a + lm];
      v2f b1; b1.x = wlds[kk * WS + n0a + 16 + lm];
              b1.y = wlds[(kk + 1) * WS + n0a + 16 + lm];
      acc0 = __builtin_amdgcn_wmma_f32_16x16x4_f32(false, a, false, b0,
                                                   (short)0, acc0, false, false);
      acc1 = __builtin_amdgcn_wmma_f32_16x16x4_f32(false, a, false, b1,
                                                   (short)0, acc1, false, false);
    }
    // Epilogue: z[:,t] = relu(acc + bp) + sib (pure stores, sib from LDS).
#pragma unroll
    for (int r = 0; r < 8; ++r) {
      int    mr   = r + 8 * hi;     // C/D layout: VGPR r -> M = r (+8 hi half)
      size_t base = ((size_t)(m0 + mr) * LL + t) * HH;
      float v0 = fmaxf(acc0[r] + bias0, 0.0f) + sS[mr * AS + n0a + lm];
      float v1 = fmaxf(acc1[r] + bias1, 0.0f) + sS[mr * AS + n0a + 16 + lm];
      z[base + n0a + lm]      = v0;
      z[base + n0a + 16 + lm] = v1;
    }
    __threadfence();                // device-scope: next step's TDM reads z
    __syncthreads();
  }
}

// ---------------------------------------------------------------------------
// Kernel 3: Wc (HxOUT f32 row-major) -> transposed f16 (OUT x H) so each
// B-fragment lane load in the output GEMM is one contiguous 32B read.
// ---------------------------------------------------------------------------
__global__ __launch_bounds__(256) void wc_convert_kernel(
    const float* __restrict__ Wc, _Float16* __restrict__ wct) {
  int id = blockIdx.x * 256 + threadIdx.x;     // id = n*256 + k
  int n  = id >> 8;
  int k  = id & 255;
  wct[id] = (_Float16)Wc[(size_t)k * OO + n];
}

// ---------------------------------------------------------------------------
// Kernel 4: logits = z @ Wc + bc  (32768 x 256 x 2048), 256 MB f32 output ->
// memory-bound (~11us write floor at 23.3 TB/s). v_wmma_f32_16x16x32_f16
// (f32 accumulate). One block per 16-row M-tile; the z tile is loaded ONCE
// and reused across all four 512-column N-chunks (z traffic 32 MB, not 128).
// ---------------------------------------------------------------------------
#define AHS 264  // f16 A-tile LDS stride (keeps 16B alignment, rotates banks)

__global__ __launch_bounds__(256) void out_gemm_kernel(
    const float* __restrict__ zf, const _Float16* __restrict__ wct,
    const float* __restrict__ bc, float* __restrict__ out) {
  __shared__ _Float16 sA[16 * AHS];

  const int tid = threadIdx.x;
  const int m0  = blockIdx.x * 16;
  for (int i = 0; i < 16; ++i)
    sA[i * AHS + tid] = (_Float16)zf[(size_t)(m0 + i) * HH + tid];
  __syncthreads();

  const int wave = tid >> 5;
  const int lane = tid & 31;
  const int hi   = lane >> 4;
  const int lm   = lane & 15;

  for (int nc = 0; nc < 4; ++nc) {
    const int nbase = nc * 512 + wave * 64;
    v8f acc[4] = {};
#pragma unroll
    for (int k0 = 0; k0 < HH; k0 += 32) {
      // A (16x32 f16): idx i -> K=k0+hi*16+i
      v8h a0 = *(const v8h*)&sA[lm * AHS + k0 + hi * 16];
      v8h a1 = *(const v8h*)&sA[lm * AHS + k0 + hi * 16 + 8];
      v16h a;
#pragma unroll
      for (int j = 0; j < 8; ++j) { a[j] = a0[j]; a[8 + j] = a1[j]; }
#pragma unroll
      for (int tt = 0; tt < 4; ++tt) {
        int n = nbase + tt * 16 + lm;
        // B (32x16 f16): b[i] = Wc[k0 + hi*16 + i][n] -> contiguous in wct
        v16h b = *(const v16h*)(wct + (size_t)n * HH + k0 + hi * 16);
        acc[tt] = __builtin_amdgcn_wmma_f32_16x16x32_f16(
            false, a, false, b, (short)0, acc[tt], false, false);
      }
    }
#pragma unroll
    for (int tt = 0; tt < 4; ++tt) {
#pragma unroll
      for (int r = 0; r < 8; ++r) {
        int row = m0 + r + 8 * hi;
        int n   = nbase + tt * 16 + lm;
        out[(size_t)row * OO + n] = acc[tt][r] + bc[n];
      }
    }
  }
}

// ---------------------------------------------------------------------------
// Launch: sib_prep -> wc_convert -> tree_recurrence -> out_gemm on `stream`.
// Workspace: z (32 MB f32) then Wc^T f16 (1 MB).
// ---------------------------------------------------------------------------
extern "C" void kernel_launch(void* const* d_in, const int* in_sizes, int n_in,
                              void* d_out, int out_size, void* d_ws, size_t ws_size,
                              hipStream_t stream) {
  const float* mol = (const float*)d_in[0];   // (512,64,256) f32
  const int*   par = (const int*)  d_in[1];   // (512,64) i32
  const float* Wp  = (const float*)d_in[2];   // (256,256) f32
  const float* bp  = (const float*)d_in[3];   // (256,) f32
  const float* Wc  = (const float*)d_in[4];   // (256,2048) f32
  const float* bc  = (const float*)d_in[5];   // (2048,) f32

  float*    z   = (float*)d_ws;                                        // 32 MB
  _Float16* wct = (_Float16*)((char*)d_ws + (size_t)BB * LL * HH * 4); //  1 MB

  sib_prep_kernel<<<dim3(BB), 256, 0, stream>>>(mol, par, z);
  wc_convert_kernel<<<(OO * HH) / 256, 256, 0, stream>>>(Wc, wct);
  tree_recurrence_kernel<<<dim3(BB / 16), 256, 0, stream>>>(Wp, bp, par, z);
  out_gemm_kernel<<<dim3((BB * LL) / 16), 256, 0, stream>>>(
      z, wct, bc, (float*)d_out);
}
